// FusionDecoder_31980326486287
// MI455X (gfx1250) — matrix operand order repaired and checked by reading, not compiled, not run on hardware
//
#include <hip/hip_runtime.h>
#include <hip/hip_bf16.h>
#include <math.h>

// Problem constants (match reference)
#define B_  16
#define L_  200
#define DW  64
#define DT  56
#define PD  8
#define T_  128          // transformer dim (= K of every GEMM)
#define H_  2
#define DK  64
#define U_  20000
#define UPAD 20032       // U_ padded to multiple of 64
#define M_  (B_*L_)      // 3200 rows
#define NEG_BIG (-4294967295.0f)

typedef __attribute__((ext_vector_type(16))) __bf16 bf16x16;
typedef __attribute__((ext_vector_type(8)))  float  floatx8;

// ---------- bf16 helpers (hi/lo split: x ~= hi + lo, ~fp32 accuracy) ----------
__device__ __forceinline__ unsigned short f32_to_bf16_rn(float x) {
  unsigned int u = __float_as_uint(x);
  unsigned int r = u + 0x7fffu + ((u >> 16) & 1u);
  return (unsigned short)(r >> 16);
}
__device__ __forceinline__ void split_hi_lo(float x, unsigned short* hi, unsigned short* lo) {
  unsigned int u  = __float_as_uint(x);
  unsigned int hb = u & 0xffff0000u;
  *hi = (unsigned short)(hb >> 16);
  float hif = __uint_as_float(hb);
  *lo = f32_to_bf16_rn(x - hif);
}

// ---------- build x = concat(word, time, pos) ; also bf16 hi/lo ----------
__global__ void build_x(const float* __restrict__ emb, const float* __restrict__ temb,
                        const float* __restrict__ pos,
                        float* __restrict__ x,
                        unsigned short* __restrict__ xhi, unsigned short* __restrict__ xlo) {
  int row = blockIdx.x;          // 0..M_-1
  int c   = threadIdx.x;         // 0..127
  int l   = row % L_;
  float v;
  if (c < DW)            v = emb [row*DW + c];
  else if (c < DW + DT)  v = temb[row*DT + (c - DW)];
  else                   v = pos [l*PD + (c - DW - DT)];
  int o = row*T_ + c;
  x[o] = v;
  unsigned short h, lw; split_hi_lo(v, &h, &lw);
  xhi[o] = h; xlo[o] = lw;
}

// ---------- transpose + hi/lo-split a (K x N) fp32 weight into (N x K) bf16 ----------
__global__ void convert_wt(const float* __restrict__ W, int K, int N,
                           unsigned short* __restrict__ Whi, unsigned short* __restrict__ Wlo) {
  int idx = blockIdx.x * blockDim.x + threadIdx.x;
  int tot = K * N;
  if (idx >= tot) return;
  int n = idx / K, k = idx - n * K;
  float v = W[(long)k * N + n];
  unsigned short h, lw; split_hi_lo(v, &h, &lw);
  Whi[idx] = h; Wlo[idx] = lw;    // Wt[n*K+k] == idx
}

// ---------- small bf16x3 WMMA GEMM: one wave per 16x16 tile ----------
__global__ __launch_bounds__(32)
void gemm_bf16x3(const unsigned short* __restrict__ Ahi, const unsigned short* __restrict__ Alo, int lda,
                 const unsigned short* __restrict__ Bthi, const unsigned short* __restrict__ Btlo, int ldb,
                 const float* __restrict__ bias,
                 float* __restrict__ C, int ldc, int K, int relu,
                 unsigned short* __restrict__ Ohi, unsigned short* __restrict__ Olo, int ldo) {
  int lane = threadIdx.x & 31;
  int tn = blockIdx.x, tm = blockIdx.y;
  int arow  = tm * 16 + (lane & 15);        // A row (M)
  int brow  = tn * 16 + (lane & 15);        // Bt row (N)
  int koffA = (lane >> 4) * 16;             // lane<16: K 0..15 ; else K 16..31
  int koffB = (lane >> 4) * 16;             // lane<16: K 0..15 ; else K 16..31

  floatx8 acc = 0.f;
  for (int k0 = 0; k0 < K; k0 += 32) {
    union { bf16x16 v; unsigned short u[16]; } aH, aL, bH, bL;
    const unsigned short* apH = Ahi  + (long)arow * lda + k0 + koffA;
    const unsigned short* apL = Alo  + (long)arow * lda + k0 + koffA;
    const unsigned short* bpH = Bthi + (long)brow * ldb + k0 + koffB;
    const unsigned short* bpL = Btlo + (long)brow * ldb + k0 + koffB;
#pragma unroll
    for (int e = 0; e < 8; ++e) {
      aH.u[e] = apH[e]; aH.u[8 + e] = apH[8 + e];
      aL.u[e] = apL[e]; aL.u[8 + e] = apL[8 + e];
    }
#pragma unroll
    for (int e = 0; e < 16; ++e) { bH.u[e] = bpH[e]; bL.u[e] = bpL[e]; }
    acc = __builtin_amdgcn_wmma_f32_16x16x32_bf16(false, aH.v, false, bH.v, (short)0, acc, false, false);
    acc = __builtin_amdgcn_wmma_f32_16x16x32_bf16(false, aH.v, false, bL.v, (short)0, acc, false, false);
    acc = __builtin_amdgcn_wmma_f32_16x16x32_bf16(false, aL.v, false, bH.v, (short)0, acc, false, false);
  }
  int n     = tn * 16 + (lane & 15);
  int mbase = tm * 16 + (lane >> 4) * 8;    // C/D: elem r -> M = mbase+r, N = n
  float bv = bias ? bias[n] : 0.f;
#pragma unroll
  for (int r = 0; r < 8; ++r) {
    float v = acc[r] + bv;
    if (relu) v = fmaxf(v, 0.f);
    C[(long)(mbase + r) * ldc + n] = v;
    if (Ohi) {
      unsigned short h, lw; split_hi_lo(v, &h, &lw);
      long o2 = (long)(mbase + r) * ldo + n;
      Ohi[o2] = h; Olo[o2] = lw;
    }
  }
}

// ---------- prediction-head GEMM: C[3200 x 20000] = A[3200x128] * Bt[N x 128]^T + bias ----------
// 8 waves / block; block tile 128(M) x 64(N); K=128 resident.
// B panel (hi+lo) staged to LDS once per block with async global->LDS copies;
// each wave computes 16x64 (4 N-subtiles) reusing one A fragment per K-chunk.
#define PBN 64
#define PBM 128
#define LSTR 144   // LDS row stride in ushorts (288B: breaks 256B bank-stride, keeps 16B align)

__global__ __launch_bounds__(256)
void gemm_pred(const unsigned short* __restrict__ Ahi, const unsigned short* __restrict__ Alo,
               const unsigned short* __restrict__ Bthi, const unsigned short* __restrict__ Btlo,
               const float* __restrict__ bias, float* __restrict__ C, int N) {
  __shared__ unsigned short lbH[PBN * LSTR];   // 18 KB
  __shared__ unsigned short lbL[PBN * LSTR];   // 18 KB
  int tid  = threadIdx.x;
  int lane = tid & 31;
  int w    = tid >> 5;
  int n0   = blockIdx.x * PBN;
  int m0   = blockIdx.y * PBM + w * 16;

  // --- stage Bt[n0..n0+63][0..127] (hi & lo) into LDS: 1024 x b128 transfers per array ---
#pragma unroll
  for (int c = 0; c < 4; ++c) {
    int i   = c * 256 + tid;          // transfer index 0..1023
    int row = i >> 4;                 // 0..63
    int seg = i & 15;                 // 16B segment within 256B row payload
    const unsigned short* gH = Bthi + (size_t)(n0 + row) * T_ + seg * 8;
    const unsigned short* gL = Btlo + (size_t)(n0 + row) * T_ + seg * 8;
    unsigned int dH = (unsigned int)(size_t)(&lbH[row * LSTR + seg * 8]);
    unsigned int dL = (unsigned int)(size_t)(&lbL[row * LSTR + seg * 8]);
    asm volatile("global_load_async_to_lds_b128 %0, %1, off"
                 :: "v"(dH), "v"((unsigned long long)(size_t)gH) : "memory");
    asm volatile("global_load_async_to_lds_b128 %0, %1, off"
                 :: "v"(dL), "v"((unsigned long long)(size_t)gL) : "memory");
  }
  asm volatile("s_wait_asynccnt 0x0" ::: "memory");
  __syncthreads();

  int koffA = (lane >> 4) * 16;
  int koffB = (lane >> 4) * 16;
  const unsigned short* arH = Ahi + (size_t)(m0 + (lane & 15)) * T_;
  const unsigned short* arL = Alo + (size_t)(m0 + (lane & 15)) * T_;

  floatx8 acc0 = 0.f, acc1 = 0.f, acc2 = 0.f, acc3 = 0.f;
#pragma unroll
  for (int k0 = 0; k0 < T_; k0 += 32) {
    union { bf16x16 v; unsigned short u[16]; } aH, aL;
#pragma unroll
    for (int e = 0; e < 8; ++e) {
      aH.u[e] = arH[k0 + koffA + e]; aH.u[8 + e] = arH[k0 + koffA + 8 + e];
      aL.u[e] = arL[k0 + koffA + e]; aL.u[8 + e] = arL[k0 + koffA + 8 + e];
    }
#pragma unroll
    for (int s = 0; s < 4; ++s) {
      union { bf16x16 v; unsigned short u[16]; } bH, bL;
      int lrow = s * 16 + (lane & 15);
      const unsigned short* pH = &lbH[lrow * LSTR + k0 + koffB];
      const unsigned short* pL = &lbL[lrow * LSTR + k0 + koffB];
#pragma unroll
      for (int e = 0; e < 16; ++e) { bH.u[e] = pH[e]; bL.u[e] = pL[e]; }
      floatx8 a = (s == 0) ? acc0 : (s == 1) ? acc1 : (s == 2) ? acc2 : acc3;
      a = __builtin_amdgcn_wmma_f32_16x16x32_bf16(false, aH.v, false, bH.v, (short)0, a, false, false);
      a = __builtin_amdgcn_wmma_f32_16x16x32_bf16(false, aH.v, false, bL.v, (short)0, a, false, false);
      a = __builtin_amdgcn_wmma_f32_16x16x32_bf16(false, aL.v, false, bH.v, (short)0, a, false, false);
      if (s == 0) acc0 = a; else if (s == 1) acc1 = a; else if (s == 2) acc2 = a; else acc3 = a;
    }
  }

  int mb = m0 + (lane >> 4) * 8;
#pragma unroll
  for (int s = 0; s < 4; ++s) {
    int n = n0 + s * 16 + (lane & 15);
    if (n < N) {
      float bv = bias[n];
      floatx8 a = (s == 0) ? acc0 : (s == 1) ? acc1 : (s == 2) ? acc2 : acc3;
#pragma unroll
      for (int r = 0; r < 8; ++r) {
        // write-once result: non-temporal, keep L2 for weight panels
        __builtin_nontemporal_store(a[r] + bv, &C[(size_t)(mb + r) * N + n]);
      }
    }
  }
}

// ---------- fused masked softmax attention (fp32 VALU), one wave per (b,h,q) ----------
__global__ __launch_bounds__(32)
void attention(const float* __restrict__ Q, const float* __restrict__ Km,
               const float* __restrict__ V, const int* __restrict__ inputs,
               float* __restrict__ O, unsigned short* __restrict__ Ohi,
               unsigned short* __restrict__ Olo) {
  int wid = blockIdx.x;
  int q = wid % L_;
  int h = (wid / L_) % H_;
  int b = wid / (L_ * H_);
  int lane = threadIdx.x;
  int row  = b * L_ + q;
  int base = row * T_ + h * DK;
  int d0 = lane * 2;                       // 64 dims / 32 lanes
  float q0 = Q[base + d0], q1 = Q[base + d0 + 1];
  bool padq = (inputs[row] == 0);
  const float invT = 1.0f / (8.0f + 1e-6f);

  float sreg[7];                            // ceil(200/32) per-lane score slots
  float mx = -__builtin_inff();
  for (int j = 0; j < L_; ++j) {
    int kb = (b * L_ + j) * T_ + h * DK;
    float p = q0 * Km[kb + d0] + q1 * Km[kb + d0 + 1];
#pragma unroll
    for (int m = 1; m < 32; m <<= 1) p += __shfl_xor(p, m, 32);
    float s = p * invT;
    if (j > q || padq) s = NEG_BIG;
    if ((j & 31) == lane) sreg[j >> 5] = s;
    mx = fmaxf(mx, s);
  }
  float sum = 0.f, o0 = 0.f, o1 = 0.f;
  for (int j = 0; j < L_; ++j) {
    float sj = __shfl(sreg[j >> 5], j & 31, 32);
    float e = expf(sj - mx);
    sum += e;
    int vb = (b * L_ + j) * T_ + h * DK;
    o0 += e * V[vb + d0];
    o1 += e * V[vb + d0 + 1];
  }
  float inv = 1.f / sum;
  o0 *= inv; o1 *= inv;
  O[base + d0] = o0; O[base + d0 + 1] = o1;
  unsigned short h0, l0, h1, l1;
  split_hi_lo(o0, &h0, &l0); split_hi_lo(o1, &h1, &l1);
  Ohi[base + d0] = h0; Olo[base + d0] = l0;
  Ohi[base + d0 + 1] = h1; Olo[base + d0 + 1] = l1;
}

// ---------- Y = LayerNorm(A + Bsum) row-wise (128 cols), one wave per row ----------
__global__ __launch_bounds__(32)
void add_ln(const float* __restrict__ A, const float* __restrict__ Bsum,
            const float* __restrict__ g, const float* __restrict__ be,
            float* __restrict__ Y, unsigned short* __restrict__ Yhi,
            unsigned short* __restrict__ Ylo) {
  int row = blockIdx.x, lane = threadIdx.x;
  float v[4];
  float s = 0.f;
#pragma unroll
  for (int i = 0; i < 4; ++i) {
    int c = lane * 4 + i;
    v[i] = A[row * T_ + c] + Bsum[row * T_ + c];
    s += v[i];
  }
#pragma unroll
  for (int m = 1; m < 32; m <<= 1) s += __shfl_xor(s, m, 32);
  float mu = s * (1.0f / T_);
  float ss = 0.f;
#pragma unroll
  for (int i = 0; i < 4; ++i) { float d = v[i] - mu; ss += d * d; }
#pragma unroll
  for (int m = 1; m < 32; m <<= 1) ss += __shfl_xor(ss, m, 32);
  float rs = rsqrtf(ss * (1.0f / T_) + 1e-5f);
#pragma unroll
  for (int i = 0; i < 4; ++i) {
    int c = lane * 4 + i;
    float y = (v[i] - mu) * rs * g[c] + be[c];
    Y[row * T_ + c] = y;
    unsigned short h, lw; split_hi_lo(y, &h, &lw);
    Yhi[row * T_ + c] = h; Ylo[row * T_ + c] = lw;
  }
}

// ---------- previous-user mask: -inf at seen ids (and id 0), one wave per (b,i) ----------
__global__ __launch_bounds__(32)
void prev_mask(const int* __restrict__ inputs, float* __restrict__ out) {
  int wid = blockIdx.x;                 // b*L + i
  int b = wid / L_, i = wid % L_;
  int lane = threadIdx.x;
  float ninf = -__builtin_inff();
  float* rowp = out + (long)wid * U_;
  if (lane == 0) rowp[0] = ninf;
  for (int j = lane; j <= i; j += 32) {
    int id = inputs[b * L_ + j];
    rowp[id] = ninf;
  }
}

extern "C" void kernel_launch(void* const* d_in, const int* in_sizes, int n_in,
                              void* d_out, int out_size, void* d_ws, size_t ws_size,
                              hipStream_t stream) {
  const float* emb    = (const float*)d_in[0];
  const float* temb   = (const float*)d_in[1];
  const int*   inputs = (const int*)  d_in[2];
  const float* pos    = (const float*)d_in[3];
  const float* W_q    = (const float*)d_in[4];
  const float* W_k    = (const float*)d_in[5];
  const float* W_v    = (const float*)d_in[6];
  const float* W_o    = (const float*)d_in[7];
  const float* lin1_w = (const float*)d_in[8];
  const float* lin1_b = (const float*)d_in[9];
  const float* lin2_w = (const float*)d_in[10];
  const float* lin2_b = (const float*)d_in[11];
  const float* ln_g   = (const float*)d_in[12];
  const float* ln_b   = (const float*)d_in[13];
  const float* pred_w = (const float*)d_in[14];
  const float* pred_b = (const float*)d_in[15];
  float* out = (float*)d_out;

  char* p = (char*)d_ws;
  auto alloc = [&](size_t bytes) -> void* {
    void* r = (void*)p;
    p += (bytes + 255) & ~(size_t)255;
    return r;
  };
  const size_t MT_f = (size_t)M_ * T_ * sizeof(float);
  const size_t MT_h = (size_t)M_ * T_ * sizeof(unsigned short);
  const size_t W_h  = (size_t)T_ * T_ * sizeof(unsigned short);
  const size_t PW_h = (size_t)UPAD * T_ * sizeof(unsigned short);  // padded to 64-row multiple

  float* x            = (float*)alloc(MT_f);
  unsigned short* xhi = (unsigned short*)alloc(MT_h);
  unsigned short* xlo = (unsigned short*)alloc(MT_h);
  unsigned short* WqH = (unsigned short*)alloc(W_h);
  unsigned short* WqL = (unsigned short*)alloc(W_h);
  unsigned short* WkH = (unsigned short*)alloc(W_h);
  unsigned short* WkL = (unsigned short*)alloc(W_h);
  unsigned short* WvH = (unsigned short*)alloc(W_h);
  unsigned short* WvL = (unsigned short*)alloc(W_h);
  unsigned short* WoH = (unsigned short*)alloc(W_h);
  unsigned short* WoL = (unsigned short*)alloc(W_h);
  unsigned short* L1H = (unsigned short*)alloc(W_h);
  unsigned short* L1L = (unsigned short*)alloc(W_h);
  unsigned short* L2H = (unsigned short*)alloc(W_h);
  unsigned short* L2L = (unsigned short*)alloc(W_h);
  unsigned short* PwH = (unsigned short*)alloc(PW_h);
  unsigned short* PwL = (unsigned short*)alloc(PW_h);
  float* Qb           = (float*)alloc(MT_f);
  float* Kb           = (float*)alloc(MT_f);
  float* Vb           = (float*)alloc(MT_f);
  float* Va           = (float*)alloc(MT_f);
  unsigned short* VaH = (unsigned short*)alloc(MT_h);
  unsigned short* VaL = (unsigned short*)alloc(MT_h);
  float* Of           = (float*)alloc(MT_f);
  float* Xn           = (float*)alloc(MT_f);
  unsigned short* XnH = (unsigned short*)alloc(MT_h);
  unsigned short* XnL = (unsigned short*)alloc(MT_h);
  float* H1           = (float*)alloc(MT_f);
  unsigned short* H1H = (unsigned short*)alloc(MT_h);
  unsigned short* H1L = (unsigned short*)alloc(MT_h);
  float* Ff           = (float*)alloc(MT_f);
  float* AO           = (float*)alloc(MT_f);
  unsigned short* AOH = (unsigned short*)alloc(MT_h);
  unsigned short* AOL = (unsigned short*)alloc(MT_h);
  (void)ws_size; (void)in_sizes; (void)n_in; (void)out_size;

  // 1. build x (concat) + bf16 split
  build_x<<<M_, 128, 0, stream>>>(emb, temb, pos, x, xhi, xlo);

  // 2. weight transpose + bf16 hi/lo split
  {
    int tot = T_ * T_;
    int nb = (tot + 255) / 256;
    convert_wt<<<nb, 256, 0, stream>>>(W_q,    T_, T_, WqH, WqL);
    convert_wt<<<nb, 256, 0, stream>>>(W_k,    T_, T_, WkH, WkL);
    convert_wt<<<nb, 256, 0, stream>>>(W_v,    T_, T_, WvH, WvL);
    convert_wt<<<nb, 256, 0, stream>>>(W_o,    T_, T_, WoH, WoL);
    convert_wt<<<nb, 256, 0, stream>>>(lin1_w, T_, T_, L1H, L1L);
    convert_wt<<<nb, 256, 0, stream>>>(lin2_w, T_, T_, L2H, L2L);
    int totp = T_ * U_;
    convert_wt<<<(totp + 255) / 256, 256, 0, stream>>>(pred_w, T_, U_, PwH, PwL);
  }

  dim3 gSmall(T_ / 16, M_ / 16);   // (8, 200)
  // 3. Q/K/V projections
  gemm_bf16x3<<<gSmall, 32, 0, stream>>>(xhi, xlo, T_, WqH, WqL, T_, nullptr, Qb, T_, T_, 0, nullptr, nullptr, 0);
  gemm_bf16x3<<<gSmall, 32, 0, stream>>>(xhi, xlo, T_, WkH, WkL, T_, nullptr, Kb, T_, T_, 0, nullptr, nullptr, 0);
  gemm_bf16x3<<<gSmall, 32, 0, stream>>>(xhi, xlo, T_, WvH, WvL, T_, nullptr, Vb, T_, T_, 0, nullptr, nullptr, 0);

  // 4. masked-softmax attention
  attention<<<B_ * H_ * L_, 32, 0, stream>>>(Qb, Kb, Vb, inputs, Va, VaH, VaL);

  // 5. output projection, 6. LN(x + O)
  gemm_bf16x3<<<gSmall, 32, 0, stream>>>(VaH, VaL, T_, WoH, WoL, T_, nullptr, Of, T_, T_, 0, nullptr, nullptr, 0);
  add_ln<<<M_, 32, 0, stream>>>(x, Of, ln_g, ln_b, Xn, XnH, XnL);

  // 7-8. FFN
  gemm_bf16x3<<<gSmall, 32, 0, stream>>>(XnH, XnL, T_, L1H, L1L, T_, lin1_b, H1, T_, T_, 1, H1H, H1L, T_);
  gemm_bf16x3<<<gSmall, 32, 0, stream>>>(H1H, H1L, T_, L2H, L2L, T_, lin2_b, Ff, T_, T_, 0, nullptr, nullptr, 0);

  // 9. LN(ffn + X)
  add_ln<<<M_, 32, 0, stream>>>(Ff, Xn, ln_g, ln_b, AO, AOH, AOL);

  // 10. prediction head: blocked WMMA GEMM with async B-panel staging to LDS
  dim3 gPred((UPAD) / PBN, M_ / PBM);   // (313, 25), 256 threads
  gemm_pred<<<gPred, 256, 0, stream>>>(AOH, AOL, PwH, PwL, pred_b, out, U_);

  // 11. previous-user -inf mask
  prev_mask<<<M_, 32, 0, stream>>>(inputs, out);
}
